// AttentionHead_15556371546904
// MI455X (gfx1250) — matrix operand order repaired and checked by reading, not compiled, not run on hardware
//
#include <hip/hip_runtime.h>

// ---------------------------------------------------------------------------
// Attention head, MI455X (gfx1250, wave32, WMMA bf16 + TDM staging).
//   Kernel 1: QKV projection  [B*T,1024] x [1024,64]x3 -> bf16 Q,K,V in d_ws
//             (bias added; softmax scale 1/8 folded into Q)
//   Kernel 2: flash attention; K blocks staged via tensor_load_to_lds (TDM),
//             V staged transposed manually; never materializes TxT scores.
// Workspace: 3 * B*T*64 bf16 = 6 MB.
// ---------------------------------------------------------------------------

typedef __attribute__((ext_vector_type(16))) __bf16          v16bf;
typedef __attribute__((ext_vector_type(8)))  __bf16          v8bf;
typedef __attribute__((ext_vector_type(8)))  float           v8f;
typedef __attribute__((ext_vector_type(8)))  unsigned short  us8;
typedef __attribute__((ext_vector_type(4)))  unsigned int    v4u;
typedef __attribute__((ext_vector_type(8)))  int             v8i;
typedef __attribute__((ext_vector_type(4)))  int             v4i;

#define B_   4
#define T_   4096
#define DIN  1024
#define DH   64

#if __has_builtin(__builtin_amdgcn_tensor_load_to_lds) && \
    __has_builtin(__builtin_amdgcn_s_wait_tensorcnt)
#define HAVE_TDM 1
#else
#define HAVE_TDM 0
#endif

static __device__ __forceinline__ unsigned short f2bf(float f) {
  unsigned u = __float_as_uint(f);
  unsigned r = (u + 0x7FFFu + ((u >> 16) & 1u)) >> 16;   // round-to-nearest-even
  return (unsigned short)r;
}

static __device__ __forceinline__ v16bf cat8(v8bf lo, v8bf hi) {
  return __builtin_shufflevector(lo, hi, 0, 1, 2, 3, 4, 5, 6, 7,
                                          8, 9, 10, 11, 12, 13, 14, 15);
}

// ---------------------------------------------------------------------------
// Kernel 1: QKV projection. Block = 256 threads (8 waves), 128 rows of x.
// Each wave computes a 16-row slice across all 192 output columns
// (cols 0-63 = K, 64-127 = Q (pre-scaled by 1/8), 128-191 = V).
// ---------------------------------------------------------------------------
#define XS_STRIDE 40   // halfwords; 80B rows: 16B-aligned chunks, bank-distinct
#define WS_STRIDE 40

__global__ __launch_bounds__(256) void qkv_proj_kernel(
    const float* __restrict__ x,
    const float* __restrict__ Wk, const float* __restrict__ bk,
    const float* __restrict__ Wq, const float* __restrict__ bq,
    const float* __restrict__ Wv, const float* __restrict__ bv,
    unsigned short* __restrict__ Kws,
    unsigned short* __restrict__ Qws,
    unsigned short* __restrict__ Vws)
{
  __shared__ unsigned short xs[128 * XS_STRIDE];   // x tile, bf16 [row][k32]
  __shared__ unsigned short ws[192 * WS_STRIDE];   // W tile transposed [n][k32]

  const int tid  = threadIdx.x;
  const int lane = tid & 31;
  const int wv   = tid >> 5;
  const int half = lane >> 4;
  const int l16  = lane & 15;
  const long rowBlk = (long)blockIdx.x * 128;

  v8f acc[12];
#pragma unroll
  for (int t = 0; t < 12; ++t)
#pragma unroll
    for (int e = 0; e < 8; ++e) acc[t][e] = 0.0f;

  for (int k0 = 0; k0 < DIN; k0 += 32) {
    __syncthreads();  // previous iteration's LDS reads done before overwrite

    // Stage x: 128 rows x 32 k, float4 global loads -> packed bf16 LDS stores.
#pragma unroll
    for (int i = 0; i < 4; ++i) {
      const int idx = tid + 256 * i;          // 0..1023 float4 chunks
      const int r   = idx >> 3;
      const int c4  = idx & 7;
      const float* src = x + (rowBlk + r) * DIN + k0 + c4 * 4;
      const float4 v = *(const float4*)src;
      if (k0 + 32 < DIN) __builtin_prefetch(src + 32, 0, 0);  // next k-slice
      const unsigned lo = (unsigned)f2bf(v.x) | ((unsigned)f2bf(v.y) << 16);
      const unsigned hi = (unsigned)f2bf(v.z) | ((unsigned)f2bf(v.w) << 16);
      *(uint2*)(xs + r * XS_STRIDE + c4 * 4) = make_uint2(lo, hi);
    }
    // Stage weights transposed: ws[n][kk] with n-major so B-frags are
    // contiguous along k.  n<64:K  64..127:Q  128..191:V
    for (int idx = tid; idx < 192 * 32; idx += 256) {
      const int n  = idx % 192;
      const int kk = idx / 192;
      const int c  = n & 63;
      const float* Wsel = (n < 64) ? Wk : (n < 128) ? Wq : Wv;
      ws[n * WS_STRIDE + kk] = f2bf(Wsel[(k0 + kk) * DH + c]);
    }
    __syncthreads();

    // A fragment (16x32 bf16): two contiguous 16B chunks per lane.
    const int arow = wv * 16 + l16;
    const v8bf alo = *(const v8bf*)(xs + arow * XS_STRIDE + 8 * half);
    const v8bf ahi = *(const v8bf*)(xs + arow * XS_STRIDE + 16 + 8 * half);
    const v16bf a = cat8(alo, ahi);

    // Group B-fragment loads (4 at a time) ahead of WMMAs so LDS traffic
    // overlaps the matrix pipe instead of serializing wait->wmma per tile.
#pragma unroll
    for (int g = 0; g < 3; ++g) {
      v16bf bb[4];
#pragma unroll
      for (int u = 0; u < 4; ++u) {
        const int n = (g * 4 + u) * 16 + l16;
        bb[u] = cat8(*(const v8bf*)(ws + n * WS_STRIDE + 8 * half),
                     *(const v8bf*)(ws + n * WS_STRIDE + 16 + 8 * half));
      }
#pragma unroll
      for (int u = 0; u < 4; ++u)
        acc[g * 4 + u] = __builtin_amdgcn_wmma_f32_16x16x32_bf16(
            false, a, false, bb[u], (short)0, acc[g * 4 + u], false, false);
    }
  }

  // Epilogue: bias (+ fold 1/sqrt(64) into Q), store bf16.
#pragma unroll
  for (int t = 0; t < 12; ++t) {
    const int c192 = t * 16 + l16;
    const int c    = c192 & 63;
    const float bias = (c192 < 64) ? bk[c] : (c192 < 128) ? bq[c] : bv[c];
    unsigned short* dst = (c192 < 64) ? Kws : (c192 < 128) ? Qws : Vws;
    const float sc = (c192 >= 64 && c192 < 128) ? 0.125f : 1.0f;
#pragma unroll
    for (int r = 0; r < 8; ++r) {
      const long row = rowBlk + wv * 16 + r + 8 * half;  // C/D layout row
      dst[row * DH + c] = f2bf((acc[t][r] + bias) * sc);
    }
  }
}

// ---------------------------------------------------------------------------
// Kernel 2: flash attention.  Block = 128 threads (4 waves) = 64 query rows.
// Wave w owns a 16-row Q tile; loop over 64-key blocks staged in LDS.
// K block staged by the Tensor Data Mover (with LDS padding reproducing the
// 72-halfword bank-friendly stride); V staged transposed by the VALU path.
// ---------------------------------------------------------------------------
#define KV_STRIDE 72   // halfwords; 144B rows: 16B-aligned, bank-distinct
#define P_STRIDE  72

__global__ __launch_bounds__(128) void flash_attn_kernel(
    const unsigned short* __restrict__ Kws,
    const unsigned short* __restrict__ Qws,
    const unsigned short* __restrict__ Vws,
    float* __restrict__ out)
{
  __shared__ unsigned short Ksh[64 * KV_STRIDE];      // [key][head]
  __shared__ unsigned short Vsh[64 * KV_STRIDE];      // transposed: [head][key]
  __shared__ unsigned short Psh[4 * 16 * P_STRIDE];   // per-wave [row][key]

  const int tid  = threadIdx.x;
  const int lane = tid & 31;
  const int wv   = tid >> 5;
  const int half = lane >> 4;
  const int l16  = lane & 15;
  const int b    = blockIdx.x / (T_ / 64);
  const int qblk = blockIdx.x % (T_ / 64);
  const long qbase = (long)b * T_ + qblk * 64 + wv * 16;

  // Q A-fragments (16x64 = two 16x32 frags), loaded once; Q pre-scaled.
  v16bf qa[2];
#pragma unroll
  for (int f = 0; f < 2; ++f) {
#pragma unroll
    for (int e = 0; e < 16; ++e) {
      const int v  = e >> 1;
      const int p  = e & 1;
      const int kk = f * 32 + ((v & 3) * 2 + (v >> 2) * 16) + 8 * half + p;
      qa[f][e] = *(const __bf16*)(Qws + (qbase + l16) * DH + kk);
    }
  }

  v8f o[4];
#pragma unroll
  for (int t = 0; t < 4; ++t)
#pragma unroll
    for (int e = 0; e < 8; ++e) o[t][e] = 0.0f;

  float m_run[8], l_run[8];
#pragma unroll
  for (int r = 0; r < 8; ++r) { m_run[r] = -INFINITY; l_run[r] = 0.0f; }

  for (int j = 0; j < T_ / 64; ++j) {
    const long kbase = (long)b * T_ + j * 64;
    __syncthreads();  // all waves done reading previous K/V block

#if HAVE_TDM
    // --- K block via Tensor Data Mover: 64x64 bf16 tile, global -> LDS.
    // pad_enable with interval=32 dwords (code 4) + amount=4 dwords (code 3)
    // inserts 16B after each 128B row => 144B LDS row stride (KV_STRIDE=72).
    if (wv == 0) {
      const unsigned lds_off = (unsigned)(unsigned long long)(void*)Ksh;
      const unsigned long long ga =
          (unsigned long long)(const void*)(Kws + kbase * DH);
      v4u g0;
      g0[0] = 1u;                                        // count=1 (valid D#)
      g0[1] = lds_off;                                   // lds_addr
      g0[2] = (unsigned)(ga & 0xFFFFFFFFu);              // global_addr[31:0]
      g0[3] = (unsigned)((ga >> 32) & 0x01FFFFFFu)       // global_addr[56:32]
              | (2u << 30);                              // type=2 (image)
      v8i g1;
      g1[0] = (int)(0x00010000u        // data_size=1 (2 bytes)
                    | (1u << 20)       // pad_enable
                    | (4u << 22)       // pad_interval: 32 dwords
                    | (3u << 25));     // pad_amount:   4 dwords
      g1[1] = (int)(64u << 16);        // tensor_dim0 = 64   [low16 @ 63:48]
      g1[2] = (int)(64u << 16);        // tensor_dim1 = 64   [low16 @ 95:80]
      g1[3] = (int)(64u << 16);        // tile_dim0   = 64   [127:112]
      g1[4] = 64;                      // tile_dim1 = 64, tile_dim2 = 0
      g1[5] = 64;                      // tensor_dim0_stride[31:0] = 64
      g1[6] = 0;                       // stride hi / dim1_stride lo
      g1[7] = 0;
      const v4i gz4 = {0, 0, 0, 0};
      const v8i gz8 = {0, 0, 0, 0, 0, 0, 0, 0};
      __builtin_amdgcn_tensor_load_to_lds(g0, g1, gz4, gz4, gz8, 0);
      __builtin_amdgcn_s_wait_tensorcnt(0);
    }
    // --- V block (transposed) by the VALU path, overlapping the TDM.
#pragma unroll
    for (int i = 0; i < 4; ++i) {
      const int c   = tid + 128 * i;
      const int row = c >> 3;
      const int seg = c & 7;
      const us8 vd = *(const us8*)(Vws + (kbase + row) * DH + seg * 8);
#pragma unroll
      for (int e = 0; e < 8; ++e)
        Vsh[(seg * 8 + e) * KV_STRIDE + row] = vd[e];
    }
#else
#pragma unroll
    for (int i = 0; i < 4; ++i) {
      const int c   = tid + 128 * i;
      const int row = c >> 3;
      const int seg = c & 7;
      const us8 kd = *(const us8*)(Kws + (kbase + row) * DH + seg * 8);
      *(us8*)(Ksh + row * KV_STRIDE + seg * 8) = kd;
      const us8 vd = *(const us8*)(Vws + (kbase + row) * DH + seg * 8);
#pragma unroll
      for (int e = 0; e < 8; ++e)
        Vsh[(seg * 8 + e) * KV_STRIDE + row] = vd[e];
    }
#endif
    __syncthreads();

    // S = Q K^T : load all 8 B-frags first, then 8 WMMAs (overlap LDS/matrix).
    v16bf kb[4][2];
#pragma unroll
    for (int t = 0; t < 4; ++t) {
      const int n = t * 16 + l16;   // key column
#pragma unroll
      for (int f = 0; f < 2; ++f)
        kb[t][f] =
            cat8(*(const v8bf*)(Ksh + n * KV_STRIDE + f * 32 + 8 * half),
                 *(const v8bf*)(Ksh + n * KV_STRIDE + f * 32 + 16 + 8 * half));
    }
    v8f s[4];
#pragma unroll
    for (int t = 0; t < 4; ++t) {
#pragma unroll
      for (int e = 0; e < 8; ++e) s[t][e] = 0.0f;
#pragma unroll
      for (int f = 0; f < 2; ++f)
        s[t] = __builtin_amdgcn_wmma_f32_16x16x32_bf16(
            false, qa[f], false, kb[t][f], (short)0, s[t], false, false);
    }

    // Online softmax (rows live across 16-lane groups; element r = row r+8*half)
#pragma unroll
    for (int r = 0; r < 8; ++r) {
      float mx = fmaxf(fmaxf(s[0][r], s[1][r]), fmaxf(s[2][r], s[3][r]));
#pragma unroll
      for (int off = 1; off <= 8; off <<= 1)
        mx = fmaxf(mx, __shfl_xor(mx, off, 32));
      const float mnew  = fmaxf(m_run[r], mx);
      const float alpha = __expf(m_run[r] - mnew);
      m_run[r] = mnew;
      l_run[r] *= alpha;
#pragma unroll
      for (int t = 0; t < 4; ++t) o[t][r] *= alpha;

      float psum = 0.0f;
#pragma unroll
      for (int t = 0; t < 4; ++t) {
        const float p = __expf(s[t][r] - mnew);
        psum += p;
        Psh[wv * 16 * P_STRIDE + (r + 8 * half) * P_STRIDE + t * 16 + l16] = f2bf(p);
      }
#pragma unroll
      for (int off = 1; off <= 8; off <<= 1)
        psum += __shfl_xor(psum, off, 32);
      l_run[r] += psum;
    }

    // O += P V : reload P as A-frags (D-layout -> A-layout via per-wave LDS).
    v16bf pa[2];
#pragma unroll
    for (int f = 0; f < 2; ++f) {
      const unsigned short* base = Psh + wv * 16 * P_STRIDE + l16 * P_STRIDE + f * 32;
      pa[f] = cat8(*(const v8bf*)(base + 8 * half),
                   *(const v8bf*)(base + 16 + 8 * half));
    }
    v16bf vb[4][2];
#pragma unroll
    for (int t = 0; t < 4; ++t) {
      const int n = t * 16 + l16;   // head column
#pragma unroll
      for (int f = 0; f < 2; ++f)
        vb[t][f] =
            cat8(*(const v8bf*)(Vsh + n * KV_STRIDE + f * 32 + 8 * half),
                 *(const v8bf*)(Vsh + n * KV_STRIDE + f * 32 + 16 + 8 * half));
    }
#pragma unroll
    for (int t = 0; t < 4; ++t)
#pragma unroll
      for (int f = 0; f < 2; ++f)
        o[t] = __builtin_amdgcn_wmma_f32_16x16x32_bf16(
            false, pa[f], false, vb[t][f], (short)0, o[t], false, false);
  }

  // Epilogue: normalize and store f32 output [B*T, 64].
#pragma unroll
  for (int r = 0; r < 8; ++r) {
    const float inv = 1.0f / l_run[r];
    const long row  = qbase + r + 8 * half;
#pragma unroll
    for (int t = 0; t < 4; ++t)
      out[row * DH + t * 16 + l16] = o[t][r] * inv;
  }
}

// ---------------------------------------------------------------------------
extern "C" void kernel_launch(void* const* d_in, const int* in_sizes, int n_in,
                              void* d_out, int out_size, void* d_ws, size_t ws_size,
                              hipStream_t stream) {
  const float* x  = (const float*)d_in[0];
  const float* Wk = (const float*)d_in[1];
  const float* bk = (const float*)d_in[2];
  const float* Wq = (const float*)d_in[3];
  const float* bq = (const float*)d_in[4];
  const float* Wv = (const float*)d_in[5];
  const float* bv = (const float*)d_in[6];
  float* out = (float*)d_out;

  const size_t tokens = (size_t)B_ * T_;
  unsigned short* Kws = (unsigned short*)d_ws;         // needs 3*tokens*64*2 = 6 MB
  unsigned short* Qws = Kws + tokens * DH;
  unsigned short* Vws = Qws + tokens * DH;

  qkv_proj_kernel<<<(int)(tokens / 128), 256, 0, stream>>>(
      x, Wk, bk, Wq, bq, Wv, bv, Kws, Qws, Vws);
  flash_attn_kernel<<<B_ * (T_ / 64), 128, 0, stream>>>(Kws, Qws, Vws, out);
}
